// SimCLR_Loss_89060441850638
// MI455X (gfx1250) — matrix operand order repaired and checked by reading, not compiled, not run on hardware
//
#include <hip/hip_runtime.h>

// ---------------- problem constants ----------------
#define BATCH   8192
#define NTOT    16384            // 2*BATCH
#define DDIM    128
#define NCHUNK  8                // column chunks
#define NSTRIP  (NTOT / 16)      // 1024 row strips of 16
#define SPW     2                // row strips per wave
#define TILES_PER_CHUNK (NSTRIP / NCHUNK) // 128 column tiles per wave
// TEMP = 0.5 -> scale = 2.0 ; fixed logsumexp max = 2.0 (|cos| <= 1)
#define K2LOG2E 2.8853900817779268f      // 2*log2(e): exp(2c-2) = exp2(c*K - K)

// LDS B-panel staging: 16 rows x 256B, +16B pad per row -> 272B row stride
#define ROWSTRIDE_H 136                   // (256+16)/2 halves
#define BUF_HALVES  (16 * ROWSTRIDE_H)    // 2176 halves = 4352 B per buffer

typedef __attribute__((ext_vector_type(16))) _Float16 v16h;
typedef __attribute__((ext_vector_type(8)))  float    v8f;
typedef __attribute__((ext_vector_type(4)))  _Float16 v4h;
typedef __attribute__((ext_vector_type(4)))  unsigned int v4u;
typedef unsigned int u32;
typedef __attribute__((ext_vector_type(4))) u32 u32x4;
typedef __attribute__((ext_vector_type(8))) int i32x8;
typedef __attribute__((ext_vector_type(4))) int i32x4;

union U16H { v16h h; v4u u[2]; };

__device__ __forceinline__ v16h ld16h(const _Float16* p0, const _Float16* p1) {
    U16H r;
    r.u[0] = *(const v4u*)p0;
    r.u[1] = *(const v4u*)p1;
    return r.h;
}
__device__ __forceinline__ v16h mk16h(v4u lo, v4u hiw) {
    U16H r; r.u[0] = lo; r.u[1] = hiw; return r.h;
}
__device__ __forceinline__ float fast_exp2(float x) {
#if __has_builtin(__builtin_amdgcn_exp2f)
    return __builtin_amdgcn_exp2f(x);     // raw v_exp_f32
#else
    return __expf(x * 0.6931471805599453f);
#endif
}

// ---- Tensor Data Mover: contiguous 4KB B panel global -> LDS with 16B pad
//      after every 256B row (bank-conflict-free ds_load_b128 afterwards).
__device__ __forceinline__ void tdm_load_tile(const _Float16* gptr, u32 lds_off) {
    unsigned long long ga = (unsigned long long)(size_t)gptr;
    u32x4 g0;
    g0.x = 1u;                                            // count=1, user desc
    g0.y = lds_off;                                       // LDS byte address
    g0.z = (u32)ga;                                       // global addr lo
    g0.w = (u32)((ga >> 32) & 0x01FFFFFFu) | (2u << 30);  // addr hi | type=2
    i32x8 g1;
    // data_size=3 (8B) | pad_enable | pad_interval=5 (64 DW=256B) | pad_amount=3 (4 DW=16B)
    g1[0] = (int)((3u << 16) | (1u << 20) | (5u << 22) | (3u << 25));
    g1[1] = (int)(512u << 16);      // tensor_dim0 = 512 (8B units) -> 4KB
    g1[2] = (int)(1u << 16);        // tensor_dim1 = 1
    g1[3] = (int)(512u << 16);      // tile_dim0 = 512
    g1[4] = (int)((1u << 16) | 1u); // tile_dim2 = 1 | tile_dim1 = 1
    g1[5] = 512;                    // tensor_dim0_stride
    g1[6] = (int)(512u << 16);      // tensor_dim1_stride lo
    g1[7] = 0;
    i32x4 g2; g2[0] = 1; g2[1] = 1; g2[2] = 512; g2[3] = (int)(1u << 16);
    i32x4 g3; g3[0] = 512; g3[1] = (int)(1u << 16); g3[2] = 1; g3[3] = 0;
#if defined(__clang_major__) && (__clang_major__ >= 23)
    i32x8 gz = {0,0,0,0,0,0,0,0};
    __builtin_amdgcn_tensor_load_to_lds(g0, g1, g2, g3, gz, 0);
#else
    __builtin_amdgcn_tensor_load_to_lds(g0, g1, g2, g3, 0);
#endif
}

// ---------------- kernel 1: row-normalize, f32 -> f16 ----------------
__global__ __launch_bounds__(256)
void simclr_normalize(const float* __restrict__ zi, const float* __restrict__ zj,
                      _Float16* __restrict__ zn) {
    const int lane = threadIdx.x & 31;
    const int row  = (blockIdx.x << 3) + (threadIdx.x >> 5);
    const float* src = (row < BATCH) ? (zi + (size_t)row * DDIM)
                                     : (zj + (size_t)(row - BATCH) * DDIM);
    float4 v = ((const float4*)src)[lane];
    float ss = v.x*v.x + v.y*v.y + v.z*v.z + v.w*v.w;
    #pragma unroll
    for (int m = 1; m < 32; m <<= 1) ss += __shfl_xor(ss, m, 32);
    float inv = 1.0f / fmaxf(sqrtf(ss), 1e-8f);
    v4h h;
    h.x = (_Float16)(v.x * inv);
    h.y = (_Float16)(v.y * inv);
    h.z = (_Float16)(v.z * inv);
    h.w = (_Float16)(v.w * inv);
    *(v4h*)(zn + (size_t)row * DDIM + lane * 4) = h;
}

// ---------------- kernel 2: WMMA Gram tiles, TDM-staged B, streaming exp-sum --
// Each wave owns SPW=2 adjacent 16-row strips; all 8 waves of a block share the
// same column chunk -> one TDM per tile feeds 16 WMMA chains.
// grid.x = NSTRIP/(8*SPW) = 64, grid.y = NCHUNK
__global__ __launch_bounds__(256)
void simclr_wmma(const _Float16* __restrict__ zn,
                 float* __restrict__ partial,   // [NTOT][NCHUNK]
                 float* __restrict__ posv) {    // [NTOT]
    __shared__ _Float16 ldsB[2 * BUF_HALVES];

    const int lane   = threadIdx.x & 31;
    const int wave   = threadIdx.x >> 5;
    const int chunk  = blockIdx.y;
    const int mlo    = lane & 15;
    const int hi     = lane >> 4;

    // wave-uniform scalars (readfirstlane -> SGPR, scalar branches for specials)
    const int strip0 = __builtin_amdgcn_readfirstlane((blockIdx.x << 4) + (wave << 1));
    const int strip1 = strip0 + 1;
    const int dt0 = strip0, dt1 = strip1;                       // diagonal tiles
    const int pt0 = strip0 ^ (NSTRIP / 2), pt1 = strip1 ^ (NSTRIP / 2); // pos tiles
    const int bm0 = strip0 << 4, bm1 = strip1 << 4;

    // --- A panels (2 x 16 rows x 128 K) resident in registers ---
    const _Float16* ar0 = zn + (size_t)(bm0 + mlo) * DDIM + hi * 8;
    const _Float16* ar1 = zn + (size_t)(bm1 + mlo) * DDIM + hi * 8;
    v16h a00 = ld16h(ar0 +  0, ar0 + 16), a01 = ld16h(ar0 + 32, ar0 + 48);
    v16h a02 = ld16h(ar0 + 64, ar0 + 80), a03 = ld16h(ar0 + 96, ar0 + 112);
    v16h a10 = ld16h(ar1 +  0, ar1 + 16), a11 = ld16h(ar1 + 32, ar1 + 48);
    v16h a12 = ld16h(ar1 + 64, ar1 + 80), a13 = ld16h(ar1 + 96, ar1 + 112);

    float psum0[8], psum1[8];
    #pragma unroll
    for (int g = 0; g < 8; ++g) { psum0[g] = 0.0f; psum1[g] = 0.0f; }
    float pv0 = 0.0f, pv1 = 0.0f;
    int   pr0 = -1,   pr1 = -1;

    const int ct0 = chunk * TILES_PER_CHUNK;
    const u32 lds_base = (u32)(size_t)&ldsB[0];

    if (wave == 0) tdm_load_tile(zn + ((size_t)ct0 << 4) * DDIM, lds_base);

    const int loff = mlo * ROWSTRIDE_H + hi * 8;    // per-lane LDS read offset

    for (int i = 0; i < TILES_PER_CHUNK; ++i) {
        const int ct = ct0 + i;
        if (wave == 0) {
            if (i + 1 < TILES_PER_CHUNK) {
                tdm_load_tile(zn + ((size_t)(ct + 1) << 4) * DDIM,
                              lds_base + (u32)(((i + 1) & 1) * BUF_HALVES * 2));
                __builtin_amdgcn_s_wait_tensorcnt(1);  // tile i done, i+1 in flight
            } else {
                __builtin_amdgcn_s_wait_tensorcnt(0);
            }
        }
        __syncthreads();                               // buf[i&1] ready

        // issue ALL eight 16B LDS reads first (distinct live regs -> partial waits)
        const _Float16* bp = &ldsB[(i & 1) * BUF_HALVES + loff];
        v4u u0 = *(const v4u*)(bp +  0), u1 = *(const v4u*)(bp +  16);
        v4u u2 = *(const v4u*)(bp + 32), u3 = *(const v4u*)(bp +  48);
        v4u u4 = *(const v4u*)(bp + 64), u5 = *(const v4u*)(bp +  80);
        v4u u6 = *(const v4u*)(bp + 96), u7 = *(const v4u*)(bp + 112);
        v16h b0 = mk16h(u0, u1), b1 = mk16h(u2, u3);
        v16h b2 = mk16h(u4, u5), b3 = mk16h(u6, u7);

        // two independent accumulation chains, interleaved for matrix-pipe ILP
        v8f c0 = {0.f,0.f,0.f,0.f,0.f,0.f,0.f,0.f};
        v8f c1 = {0.f,0.f,0.f,0.f,0.f,0.f,0.f,0.f};
        c0 = __builtin_amdgcn_wmma_f32_16x16x32_f16(false, a00, false, b0, (short)0, c0, false, false);
        c1 = __builtin_amdgcn_wmma_f32_16x16x32_f16(false, a10, false, b0, (short)0, c1, false, false);
        c0 = __builtin_amdgcn_wmma_f32_16x16x32_f16(false, a01, false, b1, (short)0, c0, false, false);
        c1 = __builtin_amdgcn_wmma_f32_16x16x32_f16(false, a11, false, b1, (short)0, c1, false, false);
        c0 = __builtin_amdgcn_wmma_f32_16x16x32_f16(false, a02, false, b2, (short)0, c0, false, false);
        c1 = __builtin_amdgcn_wmma_f32_16x16x32_f16(false, a12, false, b2, (short)0, c1, false, false);
        c0 = __builtin_amdgcn_wmma_f32_16x16x32_f16(false, a03, false, b3, (short)0, c0, false, false);
        c1 = __builtin_amdgcn_wmma_f32_16x16x32_f16(false, a13, false, b3, (short)0, c1, false, false);

        // hot epilogue: one fma + one v_exp + one add per element
        #pragma unroll
        for (int g = 0; g < 8; ++g)
            psum0[g] += fast_exp2(__builtin_fmaf(c0[g], K2LOG2E, -K2LOG2E));
        #pragma unroll
        for (int g = 0; g < 8; ++g)
            psum1[g] += fast_exp2(__builtin_fmaf(c1[g], K2LOG2E, -K2LOG2E));

        // rare special tiles: scalar (SGPR) compare + branch, <=4 of 128 iters
        if (ct == dt0) {
            #pragma unroll
            for (int g = 0; g < 8; ++g) {
                const int R = bm0 + g + (hi << 3), Cg = (ct << 4) + mlo;
                if (R == Cg) psum0[g] -= fast_exp2(__builtin_fmaf(c0[g], K2LOG2E, -K2LOG2E));
            }
        }
        if (ct == dt1) {
            #pragma unroll
            for (int g = 0; g < 8; ++g) {
                const int R = bm1 + g + (hi << 3), Cg = (ct << 4) + mlo;
                if (R == Cg) psum1[g] -= fast_exp2(__builtin_fmaf(c1[g], K2LOG2E, -K2LOG2E));
            }
        }
        if (ct == pt0) {
            #pragma unroll
            for (int g = 0; g < 8; ++g) {
                const int R = bm0 + g + (hi << 3), Cg = (ct << 4) + mlo;
                if (Cg == ((R + BATCH) & (NTOT - 1))) { pv0 = 2.0f * c0[g]; pr0 = R; }
            }
        }
        if (ct == pt1) {
            #pragma unroll
            for (int g = 0; g < 8; ++g) {
                const int R = bm1 + g + (hi << 3), Cg = (ct << 4) + mlo;
                if (Cg == ((R + BATCH) & (NTOT - 1))) { pv1 = 2.0f * c1[g]; pr1 = R; }
            }
        }

        __syncthreads();        // all reads of buf[i&1] done before TDM reuses it
    }

    // reduce each row's 16 column-lanes (xor 1,2,4,8 stays within a 16-lane half)
    #pragma unroll
    for (int g = 0; g < 8; ++g) {
        float s0 = psum0[g], s1 = psum1[g];
        #pragma unroll
        for (int m = 1; m < 16; m <<= 1) {
            s0 += __shfl_xor(s0, m, 32);
            s1 += __shfl_xor(s1, m, 32);
        }
        if (mlo == 0) {
            partial[(size_t)(bm0 + g + (hi << 3)) * NCHUNK + chunk] = s0;
            partial[(size_t)(bm1 + g + (hi << 3)) * NCHUNK + chunk] = s1;
        }
    }
    if (pr0 >= 0) posv[pr0] = pv0;   // exactly one writer per row across the grid
    if (pr1 >= 0) posv[pr1] = pv1;
}

// ---------------- kernel 3: deterministic final reduction ----------------
__global__ __launch_bounds__(256)
void simclr_reduce(const float* __restrict__ partial, const float* __restrict__ posv,
                   float* __restrict__ out) {
    __shared__ float red[256];
    const int t = threadIdx.x;
    float acc = 0.0f;
    for (int r = t; r < NTOT; r += 256) {
        float s = 0.0f;
        #pragma unroll
        for (int c = 0; c < NCHUNK; ++c) s += partial[(size_t)r * NCHUNK + c];
        acc += (__logf(s) + 2.0f) - posv[r];
    }
    red[t] = acc;
    __syncthreads();
    #pragma unroll
    for (int off = 128; off > 0; off >>= 1) {
        if (t < off) red[t] += red[t + off];
        __syncthreads();
    }
    if (t == 0) out[0] = red[0] / (float)NTOT;
}

// ---------------- launcher ----------------
extern "C" void kernel_launch(void* const* d_in, const int* in_sizes, int n_in,
                              void* d_out, int out_size, void* d_ws, size_t ws_size,
                              hipStream_t stream) {
    const float* zi = (const float*)d_in[0];
    const float* zj = (const float*)d_in[1];

    // workspace: zn f16 [NTOT*DDIM] | partial f32 [NTOT*NCHUNK] | posv f32 [NTOT]
    _Float16* zn      = (_Float16*)d_ws;
    float*    partial = (float*)((char*)d_ws + (size_t)NTOT * DDIM * sizeof(_Float16));
    float*    posv    = partial + (size_t)NTOT * NCHUNK;
    float*    out     = (float*)d_out;

    simclr_normalize<<<dim3(NTOT / 8), 256, 0, stream>>>(zi, zj, zn);
    simclr_wmma<<<dim3(NSTRIP / (8 * SPW), NCHUNK), 256, 0, stream>>>(zn, partial, posv);
    simclr_reduce<<<dim3(1), 256, 0, stream>>>(partial, posv, out);
}
